// BitSelfAttention_73924977099098
// MI455X (gfx1250) — compile-verified, matrix operand order checked by reading, not run on hardware
//
#include <hip/hip_runtime.h>

// ---------------------------------------------------------------------------
// BitSelfAttention (relative_key_query) for MI455X / gfx1250, wave32 + WMMA.
// B=4, S=1024, D=1024, H=16, HD=64.
// ---------------------------------------------------------------------------

#define Bsz 4
#define Sln 1024
#define Dm  1024
#define Hn  16
#define HD  64
#define BH  (Bsz * Hn)      // 64
#define EROWS 2048          // dist table padded 2047 -> 2048

typedef __attribute__((ext_vector_type(16))) __bf16 v16bf;
typedef __attribute__((ext_vector_type(8)))  __bf16 v8bf;
typedef __attribute__((ext_vector_type(8)))  float  v8f;

__device__ __forceinline__ v8f wmma_bf16(v16bf a, v16bf b, v8f c) {
  // D = A(16x32) * B(32x16) + C, f32 accumulate
  return __builtin_amdgcn_wmma_f32_16x16x32_bf16(
      /*neg_a=*/false, a, /*neg_b=*/false, b,
      /*c_mod=*/(short)0, c, /*reuse_a=*/false, /*reuse_b=*/false);
}

// 16-bit A-operand (and B^T-operand) tile loader, bf16 source.
// Lane t holds row m = t%16; elems 0..7 -> K = 8*half + e, elems 8..15 ->
// K = 16 + 8*half + (e-8)   (ISA 7.12.2, 16-bit A 16x32 layout).
__device__ __forceinline__ v16bf load_tileA(const __bf16* base, int ld, int lane) {
  const __bf16* row = base + (size_t)(lane & 15) * ld;
  const int half = (lane >> 4) & 1;
  v8bf lo = *(const v8bf*)(row + 8 * half);
  v8bf hi = *(const v8bf*)(row + 16 + 8 * half);
  v16bf r;
#pragma unroll
  for (int i = 0; i < 8; ++i) { r[i] = lo[i]; r[8 + i] = hi[i]; }
  return r;
}

// Same, but converting from f32 source on the fly (for the QKV GEMM).
__device__ __forceinline__ v16bf load_tileA_f32(const float* base, int ld, int lane) {
  const float* row = base + (size_t)(lane & 15) * ld;
  const int half = (lane >> 4) & 1;
  v16bf r;
#pragma unroll
  for (int i = 0; i < 8; ++i) r[i] = (__bf16)row[8 * half + i];
#pragma unroll
  for (int i = 0; i < 8; ++i) r[8 + i] = (__bf16)row[16 + 8 * half + i];
  return r;
}

__device__ __forceinline__ float rowmax16(float x) {
#pragma unroll
  for (int d = 1; d < 16; d <<= 1) x = fmaxf(x, __shfl_xor(x, d, 32));
  return x;
}
__device__ __forceinline__ float rowsum16(float x) {
#pragma unroll
  for (int d = 1; d < 16; d <<= 1) x += __shfl_xor(x, d, 32);
  return x;
}

// ---------------------------------------------------------------------------
// Kernel 0: dist_emb f32 [2047,64] -> bf16 [2048,64] (row 2047 zeroed pad)
// ---------------------------------------------------------------------------
__global__ __launch_bounds__(256) void cvt_dist_kernel(const float* __restrict__ src,
                                                       __bf16* __restrict__ dst) {
  int idx = blockIdx.x * 256 + threadIdx.x;      // grid = 512*256 = 2048*64
  int row = idx >> 6;
  dst[idx] = (row < 2047) ? (__bf16)src[idx] : (__bf16)0.0f;
}

// ---------------------------------------------------------------------------
// Kernel 1: QKV projections.  Y = X * W^T + b, X:[4096,1024] f32, W:[1024,1024].
// One wave -> 16x64 output tile. Q,K stored bf16 [bh][s][64]; V stored
// transposed bf16 [bh][64][s] so P@V B-operand loads are contiguous.
// ---------------------------------------------------------------------------
__global__ __launch_bounds__(128) void qkv_kernel(
    const float* __restrict__ X,
    const float* __restrict__ Wq, const float* __restrict__ bq,
    const float* __restrict__ Wk, const float* __restrict__ bk,
    const float* __restrict__ Wv, const float* __restrict__ bv,
    __bf16* __restrict__ q_ws, __bf16* __restrict__ k_ws,
    __bf16* __restrict__ vT_ws) {
  const int lane = threadIdx.x & 31;
  const int wave = threadIdx.x >> 5;
  const int gw   = blockIdx.x * 4 + wave;        // 0..12287
  const int mat  = gw >> 12;                     // 0:Q 1:K 2:V
  const int t    = gw & 4095;
  const int row0 = (t >> 4) * 16;                // output row tile (b*S+s)
  const int col0 = (t & 15) * 64;                // output col tile (h*64+d)

  const float* W    = (mat == 0) ? Wq : ((mat == 1) ? Wk : Wv);
  const float* bias = (mat == 0) ? bq : ((mat == 1) ? bk : bv);

  v8f acc[4];
#pragma unroll
  for (int n = 0; n < 4; ++n)
#pragma unroll
    for (int v = 0; v < 8; ++v) acc[n][v] = 0.0f;

  for (int k0 = 0; k0 < Dm; k0 += 32) {
    v16bf a = load_tileA_f32(X + (size_t)row0 * Dm + k0, Dm, lane);
#pragma unroll
    for (int n = 0; n < 4; ++n) {
      v16bf bt = load_tileA_f32(W + (size_t)(col0 + n * 16) * Dm + k0, Dm, lane);
      acc[n] = wmma_bf16(a, bt, acc[n]);
    }
  }

  const int m    = lane & 15;
  const int half = lane >> 4;
  const int r0h  = row0 + 8 * half;              // global row of acc element v=0
  const int bidx = r0h >> 10;                    // batch
  const int s0   = r0h & 1023;                   // sequence position base

#pragma unroll
  for (int n = 0; n < 4; ++n) {
    const int oc = col0 + n * 16 + m;            // output column
    const int h  = oc >> 6;
    const int d  = oc & 63;
    const float bval = bias[oc];
    if (mat == 2) {
      // V transposed: [bh][d][s]; 8 rows are contiguous in s -> one b128 store
      v8bf pv;
#pragma unroll
      for (int v = 0; v < 8; ++v) pv[v] = (__bf16)(acc[n][v] + bval);
      *(v8bf*)(vT_ws + ((size_t)(bidx * Hn + h) * HD + d) * Sln + s0) = pv;
    } else {
      __bf16* dst = (mat == 0) ? q_ws : k_ws;
#pragma unroll
      for (int v = 0; v < 8; ++v)
        dst[((size_t)(bidx * Hn + h) * Sln + (s0 + v)) * HD + d] =
            (__bf16)(acc[n][v] + bval);
    }
  }
}

// ---------------------------------------------------------------------------
// Kernel 2: fused attention with relative_key_query bias, flash-style.
// One wave owns a 16-row l-tile of one (b,h); loops r in chunks of 32.
// Per chunk: scores (4 WMMA), Pq = Q x Ewin^T (6 WMMA), Pk = K x Ewin^T
// (12 WMMA), diagonal gather via LDS, online softmax, P@V (4 WMMA).
// ---------------------------------------------------------------------------
__global__ __launch_bounds__(128) void attn_kernel(
    const __bf16* __restrict__ q_ws, const __bf16* __restrict__ k_ws,
    const __bf16* __restrict__ vT_ws, const __bf16* __restrict__ e_ws,
    const float* __restrict__ mask, float* __restrict__ out) {
  __shared__ __bf16 ldsP[4][16 * 32];    // probs bf16 (C-layout -> A-layout)
  __shared__ float  ldsPq[4][16 * 48];   // Q x Ewin^T
  __shared__ float  ldsPk[4][32 * 48];   // K x Ewin^T

  const int lane  = threadIdx.x & 31;
  const int wave  = threadIdx.x >> 5;
  const int bh    = blockIdx.x >> 4;                 // 0..63
  const int ltile = (blockIdx.x & 15) * 4 + wave;    // 0..63
  const int l0    = ltile * 16;
  const int b     = bh >> 4;
  const int h     = bh & 15;
  const int m     = lane & 15;
  const int half  = lane >> 4;

  __bf16* lp  = ldsP[wave];
  float*  lpq = ldsPq[wave];
  float*  lpk = ldsPk[wave];

  // Q tile as A-operand, both K-chunks, kept in registers for whole loop
  const __bf16* qbase = q_ws + (size_t)(bh * Sln + l0) * HD;
  v16bf aq[2];
  aq[0] = load_tileA(qbase + 0, HD, lane);
  aq[1] = load_tileA(qbase + 32, HD, lane);

  v8f o[4];
  float rowM[8], rowS[8];
#pragma unroll
  for (int n = 0; n < 4; ++n)
#pragma unroll
    for (int v = 0; v < 8; ++v) o[n][v] = 0.0f;
#pragma unroll
  for (int v = 0; v < 8; ++v) { rowM[v] = -1e30f; rowS[v] = 0.0f; }

  for (int r0 = 0; r0 < Sln; r0 += 32) {
    const __bf16* kbase = k_ws + (size_t)(bh * Sln + r0) * HD;
    if (r0 + 32 < Sln)
      __builtin_prefetch((const void*)(kbase + 32 * HD), 0, 1);  // global_prefetch

    // K tiles: reused as scores B-operand and as Pk A-operand
    v16bf kt[2][2];
#pragma unroll
    for (int jt = 0; jt < 2; ++jt)
#pragma unroll
      for (int c = 0; c < 2; ++c)
        kt[jt][c] = load_tileA(kbase + (size_t)jt * 16 * HD + c * 32, HD, lane);

    // distance-embedding window: rows mbase .. mbase+47 cover all l-r in tile
    const int mbase = l0 - r0 + 992;               // always in [0, 2000]
    const __bf16* ebase = e_ws + (size_t)mbase * HD;
    v16bf be[2][3];
#pragma unroll
    for (int c = 0; c < 2; ++c)
#pragma unroll
      for (int n = 0; n < 3; ++n)
        be[c][n] = load_tileA(ebase + (size_t)n * 16 * HD + c * 32, HD, lane);

    // ---- raw scores Q.K^T (two 16-wide column tiles) ----
    v8f sc[2];
#pragma unroll
    for (int n = 0; n < 2; ++n) {
#pragma unroll
      for (int v = 0; v < 8; ++v) sc[n][v] = 0.0f;
#pragma unroll
      for (int c = 0; c < 2; ++c) sc[n] = wmma_bf16(aq[c], kt[n][c], sc[n]);
    }

    // ---- Pq[i,w] = q[l0+i] . E[mbase+w]  (16x48) ----
    {
      v8f pq[3];
#pragma unroll
      for (int n = 0; n < 3; ++n) {
#pragma unroll
        for (int v = 0; v < 8; ++v) pq[n][v] = 0.0f;
#pragma unroll
        for (int c = 0; c < 2; ++c) pq[n] = wmma_bf16(aq[c], be[c][n], pq[n]);
      }
#pragma unroll
      for (int n = 0; n < 3; ++n)
#pragma unroll
        for (int v = 0; v < 8; ++v)
          lpq[(v + 8 * half) * 48 + n * 16 + m] = pq[n][v];
    }

    // ---- Pk[j,w] = k[r0+j] . E[mbase+w]  (32x48) ----
#pragma unroll
    for (int jt = 0; jt < 2; ++jt) {
      v8f pk[3];
#pragma unroll
      for (int n = 0; n < 3; ++n) {
#pragma unroll
        for (int v = 0; v < 8; ++v) pk[n][v] = 0.0f;
#pragma unroll
        for (int c = 0; c < 2; ++c) pk[n] = wmma_bf16(kt[jt][c], be[c][n], pk[n]);
      }
#pragma unroll
      for (int n = 0; n < 3; ++n)
#pragma unroll
        for (int v = 0; v < 8; ++v)
          lpk[(jt * 16 + v + 8 * half) * 48 + n * 16 + m] = pk[n][v];
    }

    // ---- add rel bias (diagonal gather w = 31 + i - j), scale, mask ----
    float sraw[2][8];
#pragma unroll
    for (int n = 0; n < 2; ++n) {
      const float mk = mask[(size_t)b * Sln + r0 + n * 16 + m];
#pragma unroll
      for (int v = 0; v < 8; ++v) {
        const int i = v + 8 * half;
        const int j = n * 16 + m;
        const int w = 31 + i - j;                  // in [0,46]
        sraw[n][v] = (sc[n][v] + lpq[i * 48 + w] + lpk[j * 48 + w]) * 0.125f + mk;
      }
    }

    // ---- online softmax update ----
#pragma unroll
    for (int v = 0; v < 8; ++v) {
      const float ml = rowmax16(fmaxf(sraw[0][v], sraw[1][v]));
      const float mn = fmaxf(rowM[v], ml);
      const float corr = __expf(rowM[v] - mn);
      rowM[v] = mn;
      const float p0 = __expf(sraw[0][v] - mn);
      const float p1 = __expf(sraw[1][v] - mn);
      rowS[v] = rowS[v] * corr + rowsum16(p0 + p1);
#pragma unroll
      for (int n = 0; n < 4; ++n) o[n][v] *= corr;
      const int i = v + 8 * half;
      lp[i * 32 + m]      = (__bf16)p0;            // probs -> LDS (C layout)
      lp[i * 32 + 16 + m] = (__bf16)p1;
    }

    // ---- reload probs in A layout, accumulate P @ V ----
    v16bf ap = load_tileA(lp, 32, lane);
    const __bf16* vbase = vT_ws + (size_t)bh * HD * Sln + r0;
#pragma unroll
    for (int n = 0; n < 4; ++n) {
      v16bf bv = load_tileA(vbase + (size_t)(n * 16) * Sln, Sln, lane);
      o[n] = wmma_bf16(ap, bv, o[n]);
    }
  }

  // ---- epilogue: normalize and write ctx [B,S,D] ----
  float invS[8];
#pragma unroll
  for (int v = 0; v < 8; ++v) invS[v] = 1.0f / rowS[v];
#pragma unroll
  for (int n = 0; n < 4; ++n)
#pragma unroll
    for (int v = 0; v < 8; ++v) {
      const int i = v + 8 * half;
      out[((size_t)(b * Sln + l0 + i)) * Dm + h * HD + n * 16 + m] =
          o[n][v] * invS[v];
    }
}

// ---------------------------------------------------------------------------
extern "C" void kernel_launch(void* const* d_in, const int* in_sizes, int n_in,
                              void* d_out, int out_size, void* d_ws, size_t ws_size,
                              hipStream_t stream) {
  const float* X    = (const float*)d_in[0];  // hidden_states [4,1024,1024]
  const float* mask = (const float*)d_in[1];  // attention_mask [4,1,1,1024]
  const float* Wq   = (const float*)d_in[2];
  const float* bq   = (const float*)d_in[3];
  const float* Wk   = (const float*)d_in[4];
  const float* bk   = (const float*)d_in[5];
  const float* Wv   = (const float*)d_in[6];
  const float* bv   = (const float*)d_in[7];
  const float* dist = (const float*)d_in[8];  // [2047,64]

  __bf16* q_ws  = (__bf16*)d_ws;                         // 64*1024*64 bf16
  __bf16* k_ws  = q_ws  + (size_t)BH * Sln * HD;
  __bf16* vT_ws = k_ws  + (size_t)BH * Sln * HD;
  __bf16* e_ws  = vT_ws + (size_t)BH * Sln * HD;         // [2048,64] bf16

  cvt_dist_kernel<<<512, 256, 0, stream>>>(dist, e_ws);
  qkv_kernel<<<3072, 128, 0, stream>>>(X, Wq, bq, Wk, bk, Wv, bv,
                                       q_ws, k_ws, vT_ws);
  attn_kernel<<<1024, 128, 0, stream>>>(q_ws, k_ws, vT_ws, e_ws, mask,
                                        (float*)d_out);
}
